// PolicyNetV2Torch_59828894433406
// MI455X (gfx1250) — matrix operand order repaired and checked by reading, not compiled, not run on hardware
//
#include <hip/hip_runtime.h>

// ---------------------------------------------------------------------------
// PolicyNetV2 forward for MI455X (gfx1250), wave32 + WMMA f32_16x16x32_f16.
// kernel 0: convert fp32 params -> f16 (encoder weights zero-padded K16->K32,
//           actor+value packed into one 16x256 head)
// kernel 1: per-batch-element fused entity-encoder + 2x attention + max-pool
//           (one workgroup / batch elem, all intermediates in LDS)
// kernel 2: batch MLP (ego-encoder, fc1, fc2, actor+value head) as GEMMs
// All fragment padding is compile-time (no divergent predicated loads); pair
// elements are loaded as dwords.
// ---------------------------------------------------------------------------

typedef _Float16 h16;
typedef __attribute__((ext_vector_type(16))) _Float16 v16h;
typedef __attribute__((ext_vector_type(8)))  float    v8f;

#define WMMA16(a, b, c) \
  __builtin_amdgcn_wmma_f32_16x16x32_f16(false, (a), false, (b), (short)0, (c), false, false)

#define SS    128   // entities per batch element
#define EE    64    // embed dim
#define XLD   66    // LDS row stride (f16 elems) for [128 x 64] tiles
#define PLD   132   // LDS row stride for per-wave prob strip [16 x 128]
#define OLD_  66    // LDS row stride for per-wave attn-out strip [16 x 64]
#define TLD   136   // trunk stride
#define H1LD  264   // hidden stride

union Frag {
  v16h v;
  unsigned int u[8];
};

// ---------------- WMMA fragment helpers (ISA 7.12.2 layouts) ----------------
// All element-pair (k, k+1) reads are done as one 32-bit load. Preconditions:
// ld even, k0 even, base pointer 4B aligned.

// A fragment (16x32 f16, M x K), fully in-range.
__device__ __forceinline__ v16h ldA_full(const h16* src, int ld, int m0,
                                         int k0, int lane) {
  const int half = lane >> 4;
  const int m = m0 + (lane & 15);
  const unsigned int* s32 = (const unsigned int*)src;
  const int base = (m * ld + k0) >> 1;
  Frag f;
#pragma unroll
  for (int j = 0; j < 8; ++j) {
    const int kk = (j < 4) ? (8 * half + 2 * j) : (16 + 8 * half + 2 * (j - 4));
    f.u[j] = s32[base + (kk >> 1)];
  }
  return f.v;
}

// A fragment with K=16 zero-padded to 32: in the A layout j=0..3 hold k<=14
// and j=4..7 hold k>=16, so padding is compile-time (no divergent loads).
__device__ __forceinline__ v16h ldA_pad16(const h16* src, int ld, int m0,
                                          int lane) {
  const int half = lane >> 4;
  const int m = m0 + (lane & 15);
  const unsigned int* s32 = (const unsigned int*)src;
  const int base = (m * ld) >> 1;
  Frag f;
#pragma unroll
  for (int j = 0; j < 4; ++j)
    f.u[j] = s32[base + ((8 * half + 2 * j) >> 1)];
#pragma unroll
  for (int j = 4; j < 8; ++j) f.u[j] = 0u;
  return f.v;
}

// B fragment (32x16 f16, K x N) gathered from a row-major weight matrix
// W[out][in]: B[k][n] = W[(n0+n)*ldw + k0+k]  (i.e. X @ W^T for free).
__device__ __forceinline__ v16h ldBw_full(const h16* W, int ldw, int k0,
                                          int n0, int lane) {
  const int half = lane >> 4;
  const int n = n0 + (lane & 15);
  const unsigned int* s32 = (const unsigned int*)W;
  const int base = (n * ldw + k0) >> 1;
  Frag f;
#pragma unroll
  for (int j = 0; j < 8; ++j) {
    const int kk = (j < 4) ? (8 * half + 2 * j) : (16 + 8 * half + 2 * (j - 4));
    f.u[j] = s32[base + (kk >> 1)];
  }
  return f.v;
}

// B fragment gathered from a row-major matrix M[k][n] (e.g. V: B[k=t][n=d]).
__device__ __forceinline__ v16h ldBm(const h16* M, int ld, int k0, int n0,
                                     int lane) {
  const int half = lane >> 4;
  const int n = n0 + (lane & 15);
  v16h b;
#pragma unroll
  for (int j = 0; j < 8; ++j) {
    const int k = k0 + ((j < 4) ? (8 * half + 2 * j) : (16 + 8 * half + 2 * (j - 4)));
    b[2 * j]     = M[k * ld + n];
    b[2 * j + 1] = M[(k + 1) * ld + n];
  }
  return b;
}

// Store C/D fragment (16x16 f32) to row-major f16 with optional relu.
__device__ __forceinline__ void stC(h16* dst, int ld, int m0, int n0,
                                    const v8f& c, int lane, bool relu) {
  const int half = lane >> 4;
  const int n = n0 + (lane & 15);
#pragma unroll
  for (int r = 0; r < 8; ++r) {
    float v = c[r];
    if (relu) v = fmaxf(v, 0.0f);
    dst[(m0 + r + 8 * half) * ld + n] = (h16)v;
  }
}

__device__ __forceinline__ v8f biasC(const float* bias, int n0, int lane) {
  const float bv = bias[n0 + (lane & 15)];
  v8f c;
#pragma unroll
  for (int r = 0; r < 8; ++r) c[r] = bv;
  return c;
}

__device__ __forceinline__ v8f zeroC() {
  v8f c;
#pragma unroll
  for (int r = 0; r < 8; ++r) c[r] = 0.0f;
  return c;
}

// One 16-row x 64-col projection: dst = a(16x64) @ W^T + bias.
__device__ __forceinline__ void proj64(const v16h& a0, const v16h& a1,
                                       const h16* __restrict__ W,
                                       const float* __restrict__ bias,
                                       h16* dst, int m0, int lane) {
#pragma unroll
  for (int t = 0; t < 4; ++t) {
    v8f c = biasC(bias, t * 16, lane);
    v16h b0 = ldBw_full(W, 64, 0, t * 16, lane);
    c = WMMA16(a0, b0, c);
    v16h b1 = ldBw_full(W, 64, 32, t * 16, lane);
    c = WMMA16(a1, b1, c);
    stC(dst, XLD, m0, t * 16, c, lane, false);
  }
}

// ---------------------------- kernel 0: prep -------------------------------

struct PrepArgs {
  const float* src[12];
  h16*         dst[12];
  int          n[12];
  const float* wa;   // actor weight [9,256]
  const float* wvp;  // value weight [1,256]
  const float* ba;   // actor bias [9]
  const float* bvv;  // value bias [1]
  h16*   o_whead;    // [16,256] packed head weight (rows 0-8 actor, 9 value)
  float* o_bhead;    // [16]
};

__global__ __launch_bounds__(256) void prep_kernel(PrepArgs p) {
  const int seg = blockIdx.x;
  const int tid = threadIdx.x;
  if (seg < 2) {
    // encoder weights [64,16] fp32 -> [64,32] f16 zero-padded
    const float* s = p.src[seg];
    h16* d = p.dst[seg];
    for (int i = tid; i < 64 * 32; i += 256) {
      const int n = i >> 5, k = i & 31;
      d[i] = (k < 16) ? (h16)s[n * 16 + k] : (h16)0.0f;
    }
  } else if (seg < 12) {
    const float* s = p.src[seg];
    h16* d = p.dst[seg];
    const int n = p.n[seg];
    for (int i = tid; i < n; i += 256) d[i] = (h16)s[i];
  } else {
    for (int i = tid; i < 16 * 256; i += 256) {
      const int n = i >> 8, k = i & 255;
      float v = 0.0f;
      if (n < 9)       v = p.wa[n * 256 + k];
      else if (n == 9) v = p.wvp[k];
      p.o_whead[i] = (h16)v;
    }
    if (tid < 16)
      p.o_bhead[tid] = (tid < 9) ? p.ba[tid] : ((tid == 9) ? p.bvv[0] : 0.0f);
  }
}

// ------------------------ kernel 1: fused attention -------------------------

struct AttnArgs {
  const float* ent;   // [B,128,16]
  const int*   nent;  // [B]
  const h16*   went;  // [64,32] f16 (zero-padded)
  const float* bent;  // [64]
  const h16 *wq[2], *wk[2], *wv[2], *wo[2];        // [64,64] f16
  const float *bq[2], *bk[2], *bv[2], *bo[2];      // [64]
  const float *lng[2], *lnb[2];                    // [64]
  h16* pooled;  // [B,64] f16 out
};

__global__ __launch_bounds__(256) void attn_kernel(AttnArgs p) {
  __shared__ alignas(16) h16 sX[SS * XLD];        // residual stream
  __shared__ alignas(16) h16 sQ[SS * XLD];        // also entity staging
  __shared__ alignas(16) h16 sK[SS * XLD + 16];   // +16 zero tail (B over-read)
  __shared__ alignas(16) h16 sV[SS * XLD];
  __shared__ alignas(16) h16 sP[8 * 16 * PLD];    // per-wave prob strips
  __shared__ alignas(16) h16 sO[8 * 16 * OLD_];   // per-wave attn-out strips

  const int b    = blockIdx.x;
  const int tid  = threadIdx.x;
  const int lane = tid & 31;
  const int wv   = tid >> 5;         // wave id 0..7 (row-block owner)
  const int half = lane >> 4;
  const int ncol = lane & 15;
  const int m0   = wv * 16;
  const int nent = p.nent[b];

  // ---- zero-init Q/K/V tiles so all padding/over-read columns are finite ----
  {
    unsigned int* zq = (unsigned int*)sQ;
    unsigned int* zk = (unsigned int*)sK;
    unsigned int* zv = (unsigned int*)sV;
    for (int i = tid; i < (SS * XLD) / 2; i += 256) {
      zq[i] = 0u; zk[i] = 0u; zv[i] = 0u;
    }
    if (tid < 8) zk[(SS * XLD) / 2 + tid] = 0u;
  }
  __syncthreads();

  // ---- stage entities (fp32 -> f16) into sQ[:, 0:16] ----
  {
    const float* eb = p.ent + (size_t)b * SS * 16;
    for (int i = tid; i < SS * 16; i += 256)
      sQ[(i >> 4) * XLD + (i & 15)] = (h16)eb[i];
  }
  __syncthreads();

  // ---- entity encoder: relu(ent @ Went^T + bent) -> sX ----
  {
    const v16h a = ldA_pad16(sQ, XLD, m0, lane);  // K=16 padded to 32
#pragma unroll
    for (int t = 0; t < 4; ++t) {
      v8f c = biasC(p.bent, t * 16, lane);
      const v16h bf = ldBw_full(p.went, 32, 0, t * 16, lane);  // zero-padded W
      c = WMMA16(a, bf, c);
      stC(sX, XLD, m0, t * 16, c, lane, true);
    }
  }
  __syncthreads();

  for (int layer = 0; layer < 2; ++layer) {
    // ---- phase 1: Q/K/V projections for this wave's 16-row strip ----
    {
      const v16h a0 = ldA_full(sX, XLD, m0, 0, lane);
      const v16h a1 = ldA_full(sX, XLD, m0, 32, lane);
      proj64(a0, a1, p.wq[layer], p.bq[layer], sQ, m0, lane);
      proj64(a0, a1, p.wk[layer], p.bk[layer], sK, m0, lane);
      proj64(a0, a1, p.wv[layer], p.bv[layer], sV, m0, lane);
    }
    __syncthreads();

    // ---- phase 2: per-head scores -> softmax -> attn @ V ----
    h16* myP = sP + wv * 16 * PLD;
    h16* myO = sO + wv * 16 * OLD_;
    for (int h = 0; h < 4; ++h) {
      const int hb = h * 16;
      // Q slice: exact zeros for k>=16, so the K-side may read finite garbage
      const v16h aq = ldA_pad16(sQ + hb, XLD, m0, lane);
      v8f sc[8];
#pragma unroll
      for (int t = 0; t < 8; ++t) {
        const v16h bk = ldBw_full(sK + hb, XLD, 0, t * 16, lane);  // K^T gather
        v8f c = zeroC();
        c = WMMA16(aq, bk, c);
        sc[t] = c;
      }
      // scale (D^-0.5 = 0.25), clip +-50, key mask, row softmax, clamp 1e-8
#pragma unroll
      for (int r = 0; r < 8; ++r) {
        float mx = -1e30f;
#pragma unroll
        for (int t = 0; t < 8; ++t) {
          float v = sc[t][r] * 0.25f;
          v = fminf(fmaxf(v, -50.0f), 50.0f);
          if ((t * 16 + ncol) >= nent) v = -10000.0f;
          sc[t][r] = v;
          mx = fmaxf(mx, v);
        }
#pragma unroll
        for (int msk = 1; msk < 16; msk <<= 1)
          mx = fmaxf(mx, __shfl_xor(mx, msk, 32));
        float sum = 0.0f;
#pragma unroll
        for (int t = 0; t < 8; ++t) {
          const float e = __expf(sc[t][r] - mx);
          sc[t][r] = e;
          sum += e;
        }
#pragma unroll
        for (int msk = 1; msk < 16; msk <<= 1)
          sum += __shfl_xor(sum, msk, 32);
        const float inv = 1.0f / sum;
        const int mrow = r + 8 * half;
#pragma unroll
        for (int t = 0; t < 8; ++t)
          myP[mrow * PLD + t * 16 + ncol] = (h16)fmaxf(sc[t][r] * inv, 1e-8f);
      }
      // out_h = probs @ V_h   (K = 128 over keys)
      v8f oc = zeroC();
#pragma unroll
      for (int kb = 0; kb < 4; ++kb) {
        const v16h ap = ldA_full(myP, PLD, 0, kb * 32, lane);
        const v16h bv = ldBm(sV + hb, XLD, kb * 32, 0, lane);
        oc = WMMA16(ap, bv, oc);
      }
      stC(myO, OLD_, 0, hb, oc, lane, false);
    }

    // ---- phase 3: output projection + residual + layernorm (wave-local) ----
    {
      const v16h a0 = ldA_full(myO, OLD_, 0, 0, lane);
      const v16h a1 = ldA_full(myO, OLD_, 0, 32, lane);
      v8f co[4];
#pragma unroll
      for (int t = 0; t < 4; ++t) {
        v8f c = biasC(p.bo[layer], t * 16, lane);
        v16h b0 = ldBw_full(p.wo[layer], 64, 0, t * 16, lane);
        c = WMMA16(a0, b0, c);
        v16h b1 = ldBw_full(p.wo[layer], 64, 32, t * 16, lane);
        c = WMMA16(a1, b1, c);
        co[t] = c;
      }
      float gl[4], bl[4];
#pragma unroll
      for (int t = 0; t < 4; ++t) {
        gl[t] = p.lng[layer][t * 16 + ncol];
        bl[t] = p.lnb[layer][t * 16 + ncol];
      }
#pragma unroll
      for (int r = 0; r < 8; ++r) {
        const int m = m0 + r + 8 * half;
        float x4[4], sum = 0.0f, sq = 0.0f;
#pragma unroll
        for (int t = 0; t < 4; ++t) {
          const float v = co[t][r] + (float)sX[m * XLD + t * 16 + ncol];
          x4[t] = v;
          sum += v;
          sq += v * v;
        }
#pragma unroll
        for (int msk = 1; msk < 16; msk <<= 1) {
          sum += __shfl_xor(sum, msk, 32);
          sq  += __shfl_xor(sq, msk, 32);
        }
        const float mu  = sum * (1.0f / 64.0f);
        const float var = sq * (1.0f / 64.0f) - mu * mu;
        const float inv = rsqrtf(var + 1e-5f);
#pragma unroll
        for (int t = 0; t < 4; ++t)
          sX[m * XLD + t * 16 + ncol] = (h16)((x4[t] - mu) * inv * gl[t] + bl[t]);
      }
    }
    __syncthreads();
  }

  // ---- masked max-pool over valid entities ----
  if (tid < 64) {
    float mx = -1e30f;
    for (int m = 0; m < nent; ++m)
      mx = fmaxf(mx, (float)sX[m * XLD + tid]);
    if (nent == 0) mx = 0.0f;
    p.pooled[(size_t)b * 64 + tid] = (h16)mx;
  }
}

// ----------------------------- kernel 2: MLP -------------------------------

struct MlpArgs {
  const float* ego;     // [B,16]
  const h16*   wego;    // [64,32] f16 (zero-padded)
  const float* bego;    // [64]
  const h16*   pooled;  // [B,64] f16
  const h16*   w1;      // [256,128] f16
  const float* b1;      // [256]
  const h16*   w2;      // [256,256] f16
  const float* b2;      // [256]
  const h16*   whead;   // [16,256] f16 (actor rows 0-8, value row 9)
  const float* bhead;   // [16]
  float*       out;     // logits [B,9] then values [B]
};

__global__ __launch_bounds__(256) void mlp_kernel(MlpArgs p) {
  __shared__ alignas(16) h16 sT[16 * TLD];    // trunk [16 x 128]
  __shared__ alignas(16) h16 sH1[16 * H1LD];  // [16 x 256]
  __shared__ alignas(16) h16 sH2[16 * H1LD];  // [16 x 256]
  __shared__ alignas(16) h16 sEgo[16 * 16];

  const int tid = threadIdx.x, lane = tid & 31, wv = tid >> 5;
  const int r0 = blockIdx.x * 16;

  __builtin_prefetch(p.w2, 0, 3);  // global_prefetch of the big weight

  for (int i = tid; i < 16 * 16; i += 256)
    sEgo[i] = (h16)p.ego[(size_t)(r0 + (i >> 4)) * 16 + (i & 15)];
  for (int i = tid; i < 16 * 64; i += 256) {
    const int m = i >> 6, c = i & 63;
    sT[m * TLD + 64 + c] = p.pooled[(size_t)(r0 + m) * 64 + c];
  }
  __syncthreads();

  // ego encoder: relu(ego @ Wego^T + b) -> trunk cols 0..63 (waves 0-3)
  if (wv < 4) {
    const v16h a = ldA_pad16(sEgo, 16, 0, lane);
    v8f c = biasC(p.bego, wv * 16, lane);
    const v16h bf = ldBw_full(p.wego, 32, 0, wv * 16, lane);
    c = WMMA16(a, bf, c);
    stC(sT, TLD, 0, wv * 16, c, lane, true);
  }
  __syncthreads();

  // fc1: [16,128] @ [128,256]
  {
    v16h a[4];
#pragma unroll
    for (int kb = 0; kb < 4; ++kb) a[kb] = ldA_full(sT, TLD, 0, kb * 32, lane);
    for (int tt = wv; tt < 16; tt += 8) {
      v8f c = biasC(p.b1, tt * 16, lane);
#pragma unroll
      for (int kb = 0; kb < 4; ++kb) {
        const v16h bf = ldBw_full(p.w1, 128, kb * 32, tt * 16, lane);
        c = WMMA16(a[kb], bf, c);
      }
      stC(sH1, H1LD, 0, tt * 16, c, lane, true);
    }
  }
  __syncthreads();

  // fc2: [16,256] @ [256,256]
  {
    v16h a[8];
#pragma unroll
    for (int kb = 0; kb < 8; ++kb) a[kb] = ldA_full(sH1, H1LD, 0, kb * 32, lane);
    for (int tt = wv; tt < 16; tt += 8) {
      v8f c = biasC(p.b2, tt * 16, lane);
#pragma unroll
      for (int kb = 0; kb < 8; ++kb) {
        const v16h bf = ldBw_full(p.w2, 256, kb * 32, tt * 16, lane);
        c = WMMA16(a[kb], bf, c);
      }
      stC(sH2, H1LD, 0, tt * 16, c, lane, true);
    }
  }
  __syncthreads();

  // head: [16,256] @ [256,16] -> logits (n<9) + value (n==9)
  if (wv == 0) {
    v8f c = biasC(p.bhead, 0, lane);
#pragma unroll
    for (int kb = 0; kb < 8; ++kb) {
      const v16h a = ldA_full(sH2, H1LD, 0, kb * 32, lane);
      const v16h bf = ldBw_full(p.whead, 256, kb * 32, 0, lane);
      c = WMMA16(a, bf, c);
    }
    const int n = lane & 15, half = lane >> 4;
#pragma unroll
    for (int r = 0; r < 8; ++r) {
      const int m = r0 + r + 8 * half;
      const float v = c[r];
      if (n < 9)       p.out[(size_t)m * 9 + n] = v;
      else if (n == 9) p.out[1024 * 9 + m] = v;
    }
  }
}

// ------------------------------- launch ------------------------------------

// d_in order (setup_inputs insertion order, params flattened leaf-by-leaf):
enum {
  IN_EGO = 0, IN_ENT, IN_NENT,
  IN_WEGO, IN_BEGO, IN_WENT, IN_BENT,
  IN_A1_WQ, IN_A1_BQ, IN_A1_WK, IN_A1_BK, IN_A1_WV, IN_A1_BV,
  IN_A1_WO, IN_A1_BO, IN_A1_LNG, IN_A1_LNB,
  IN_A2_WQ, IN_A2_BQ, IN_A2_WK, IN_A2_BK, IN_A2_WV, IN_A2_BV,
  IN_A2_WO, IN_A2_BO, IN_A2_LNG, IN_A2_LNB,
  IN_FC1_W, IN_FC1_B, IN_FC2_W, IN_FC2_B,
  IN_ACT_W, IN_ACT_B, IN_VAL_W, IN_VAL_B
};

extern "C" void kernel_launch(void* const* d_in, const int* in_sizes, int n_in,
                              void* d_out, int out_size, void* d_ws, size_t ws_size,
                              hipStream_t stream) {
  (void)in_sizes; (void)n_in; (void)out_size; (void)ws_size;
  char* ws = (char*)d_ws;

  // workspace layout (bytes) — total ~410 KB
  h16* o_wego = (h16*)(ws + 0);      // [64,32] padded: 4096 B
  h16* o_went = (h16*)(ws + 4096);   // [64,32] padded: 4096 B
  size_t base = 8192;
  h16 *o_wq[2], *o_wk[2], *o_wv[2], *o_wo[2];
  for (int l = 0; l < 2; ++l) {
    o_wq[l] = (h16*)(ws + base); base += 8192;
    o_wk[l] = (h16*)(ws + base); base += 8192;
    o_wv[l] = (h16*)(ws + base); base += 8192;
    o_wo[l] = (h16*)(ws + base); base += 8192;
  }
  h16*   o_w1   = (h16*)(ws + base);   base += 65536;
  h16*   o_w2   = (h16*)(ws + base);   base += 131072;
  h16*   o_wh   = (h16*)(ws + base);   base += 8192;
  float* o_bh   = (float*)(ws + base); base += 256;
  h16*   o_pool = (h16*)(ws + base);   base += 131072;

  // ---- kernel 0: param prep ----
  PrepArgs pp;
  const int srcIdx[12] = {IN_WEGO, IN_WENT,
                          IN_A1_WQ, IN_A1_WK, IN_A1_WV, IN_A1_WO,
                          IN_A2_WQ, IN_A2_WK, IN_A2_WV, IN_A2_WO,
                          IN_FC1_W, IN_FC2_W};
  h16* dsts[12] = {o_wego, o_went,
                   o_wq[0], o_wk[0], o_wv[0], o_wo[0],
                   o_wq[1], o_wk[1], o_wv[1], o_wo[1],
                   o_w1, o_w2};
  const int cnts[12] = {2048, 2048, 4096, 4096, 4096, 4096,
                        4096, 4096, 4096, 4096, 32768, 65536};
  for (int i = 0; i < 12; ++i) {
    pp.src[i] = (const float*)d_in[srcIdx[i]];
    pp.dst[i] = dsts[i];
    pp.n[i]   = cnts[i];
  }
  pp.wa = (const float*)d_in[IN_ACT_W];
  pp.wvp = (const float*)d_in[IN_VAL_W];
  pp.ba = (const float*)d_in[IN_ACT_B];
  pp.bvv = (const float*)d_in[IN_VAL_B];
  pp.o_whead = o_wh;
  pp.o_bhead = o_bh;
  prep_kernel<<<13, 256, 0, stream>>>(pp);

  // ---- kernel 1: fused attention over entities ----
  AttnArgs aa;
  aa.ent  = (const float*)d_in[IN_ENT];
  aa.nent = (const int*)d_in[IN_NENT];
  aa.went = o_went;
  aa.bent = (const float*)d_in[IN_BENT];
  aa.wq[0] = o_wq[0]; aa.wk[0] = o_wk[0]; aa.wv[0] = o_wv[0]; aa.wo[0] = o_wo[0];
  aa.wq[1] = o_wq[1]; aa.wk[1] = o_wk[1]; aa.wv[1] = o_wv[1]; aa.wo[1] = o_wo[1];
  aa.bq[0] = (const float*)d_in[IN_A1_BQ]; aa.bk[0] = (const float*)d_in[IN_A1_BK];
  aa.bv[0] = (const float*)d_in[IN_A1_BV]; aa.bo[0] = (const float*)d_in[IN_A1_BO];
  aa.lng[0] = (const float*)d_in[IN_A1_LNG]; aa.lnb[0] = (const float*)d_in[IN_A1_LNB];
  aa.bq[1] = (const float*)d_in[IN_A2_BQ]; aa.bk[1] = (const float*)d_in[IN_A2_BK];
  aa.bv[1] = (const float*)d_in[IN_A2_BV]; aa.bo[1] = (const float*)d_in[IN_A2_BO];
  aa.lng[1] = (const float*)d_in[IN_A2_LNG]; aa.lnb[1] = (const float*)d_in[IN_A2_LNB];
  aa.pooled = o_pool;
  attn_kernel<<<1024, 256, 0, stream>>>(aa);

  // ---- kernel 2: batch MLP ----
  MlpArgs mm;
  mm.ego    = (const float*)d_in[IN_EGO];
  mm.wego   = o_wego;
  mm.bego   = (const float*)d_in[IN_BEGO];
  mm.pooled = o_pool;
  mm.w1 = o_w1; mm.b1 = (const float*)d_in[IN_FC1_B];
  mm.w2 = o_w2; mm.b2 = (const float*)d_in[IN_FC2_B];
  mm.whead = o_wh; mm.bhead = o_bh;
  mm.out = (float*)d_out;
  mlp_kernel<<<64, 256, 0, stream>>>(mm);
}
